// HybridAttention_38122129719911
// MI455X (gfx1250) — matrix operand order repaired and checked by reading, not compiled, not run on hardware
//
#include <hip/hip_runtime.h>
#include <hip/hip_bf16.h>

// ---------------- problem constants ----------------
#define B_    2
#define L_    2048
#define HID_  2048
#define NH_   16
#define NKV_  8
#define HD_   128
#define NIMG_ 1024

typedef __attribute__((ext_vector_type(16))) __bf16 v16bf;
typedef __attribute__((ext_vector_type(8)))  float  v8f;

union FragU { v16bf v; uint4 q[2]; };

// A-operand (MxK, 16x32 bf16): lane l -> m = l%16, h = l/16
// element i<8 : K = k0 + 8h + i ; element i>=8 : K = k0 + 16 + 8h + (i-8)
__device__ __forceinline__ v16bf lds_load_a(const __bf16* base, int row, int ld, int k0) {
  const int lane = threadIdx.x & 31;
  const int m = lane & 15, h = lane >> 4;
  const __bf16* p = base + (size_t)(row + m) * ld + k0;
  FragU f;
  f.q[0] = *(const uint4*)(p + 8 * h);
  f.q[1] = *(const uint4*)(p + 16 + 8 * h);
  return f.v;
}

// B-operand (KxN, 32x16 bf16) stored as N-major rows with K contiguous:
// lane l -> n = l%16, h = l/16 ; element i<8 : K = k0 + 8h + i ; element i>=8 : K = k0 + 16 + 8h + (i-8)
__device__ __forceinline__ v16bf lds_load_b(const __bf16* base, int col, int ld, int k0) {
  const int lane = threadIdx.x & 31;
  const int n = lane & 15, h = lane >> 4;
  const __bf16* p = base + (size_t)(col + n) * ld + k0;
  FragU f;
  f.q[0] = *(const uint4*)(p + 8 * h);
  f.q[1] = *(const uint4*)(p + 16 + 8 * h);
  return f.v;
}

// CDNA5 async global->LDS copy (16B per lane), tracked by ASYNCcnt.
// vdst = per-lane LDS byte address (low 32 bits of generic shared pointer),
// vaddr = 64-bit global address, saddr = off.
__device__ __forceinline__ void async_copy16(void* lds, const void* gptr) {
  asm volatile("global_load_async_to_lds_b128 %0, %1, off"
               :: "v"((unsigned int)(size_t)lds), "v"(gptr)
               : "memory");
}
#define S_WAIT_ASYNC(immstr) asm volatile("s_wait_asynccnt " immstr ::: "memory")

// ---------------- fp32 -> bf16 conversion ----------------
struct __align__(8) bf4 { __bf16 x, y, z, w; };

__global__ __launch_bounds__(256) void cvt_f32_bf16(const float4* __restrict__ src,
                                                    bf4* __restrict__ dst, int nvec) {
  int i = blockIdx.x * 256 + threadIdx.x;
  if (i < nvec) {
    float4 f = src[i];
    bf4 o;
    o.x = (__bf16)f.x; o.y = (__bf16)f.y; o.z = (__bf16)f.z; o.w = (__bf16)f.w;
    dst[i] = o;
  }
}

// ---------------- fused QKV GEMM + RMSNorm + RoPE ----------------
// C[m,n] = sum_k xb[m,k]*wb[n,k]; M=4096, N=4096(Q|K|V), K=2048
// tile 64x128, BK=64, double-buffered async LDS staging; 8 waves (2x4),
// per-wave 32x32 via 2x2 frags x 2 k-steps.
__global__ __launch_bounds__(256) void qkv_gemm_kernel(
    const __bf16* __restrict__ xb, const __bf16* __restrict__ wb,
    const float* __restrict__ cosp, const float* __restrict__ sinp,
    const float* __restrict__ qw, const float* __restrict__ kw,
    __bf16* __restrict__ Qo, __bf16* __restrict__ Ko, __bf16* __restrict__ Vto) {
  __shared__ __align__(16) __bf16 As[2][64][72];
  __shared__ __align__(16) __bf16 Bs[2][128][72];
  __shared__ __align__(16) float  Cs[64][132];

  const int tid = threadIdx.x;
  const int lane = tid & 31;
  const int w  = tid >> 5;
  const int wr = w >> 2;   // 0..1 (32 rows)
  const int wc = w & 3;    // 0..3 (32 cols)
  const int hh = lane >> 4, nn = lane & 15;

  const int m0 = blockIdx.y * 64;
  const int n0 = blockIdx.x * 128;

  v8f acc[2][2] = {};

  // async stage one BK=64 tile: A = 2 chunks/thread, B = 4 chunks/thread (6 total)
  auto issue_tile = [&](int buf, int k0) {
#pragma unroll
    for (int j = 0; j < 2; ++j) {
      int idx = tid + 256 * j;
      int r = idx >> 3, c = (idx & 7) * 8;
      async_copy16(&As[buf][r][c], xb + (size_t)(m0 + r) * HID_ + k0 + c);
    }
#pragma unroll
    for (int j = 0; j < 4; ++j) {
      int idx = tid + 256 * j;
      int r = idx >> 3, c = (idx & 7) * 8;
      async_copy16(&Bs[buf][r][c], wb + (size_t)(n0 + r) * HID_ + k0 + c);
    }
  };

  issue_tile(0, 0);
  const int NT = HID_ / 64;   // 32 k-tiles
  for (int t = 0; t < NT; ++t) {
    if (t + 1 < NT) {
      issue_tile((t + 1) & 1, (t + 1) * 64);
      S_WAIT_ASYNC("0x6");          // newest 6 (next tile) may remain in flight
    } else {
      S_WAIT_ASYNC("0x0");
    }
    __syncthreads();                 // tile t resident for all waves
    const int bi = t & 1;
#pragma unroll
    for (int kk = 0; kk < 64; kk += 32) {
      v16bf afr[2], bfr[2];
#pragma unroll
      for (int mi = 0; mi < 2; ++mi)
        afr[mi] = lds_load_a(&As[bi][0][0], wr * 32 + mi * 16, 72, kk);
#pragma unroll
      for (int ni = 0; ni < 2; ++ni)
        bfr[ni] = lds_load_b(&Bs[bi][0][0], wc * 32 + ni * 16, 72, kk);
#pragma unroll
      for (int mi = 0; mi < 2; ++mi)
#pragma unroll
        for (int ni = 0; ni < 2; ++ni)
          acc[mi][ni] = __builtin_amdgcn_wmma_f32_16x16x32_bf16(
              false, afr[mi], false, bfr[ni], (short)0, acc[mi][ni], false, false);
    }
    __syncthreads();                 // done reading buf bi before it is re-filled
  }

  // spill C tile to LDS for row-wise epilogue
#pragma unroll
  for (int mi = 0; mi < 2; ++mi)
#pragma unroll
    for (int ni = 0; ni < 2; ++ni)
#pragma unroll
      for (int e = 0; e < 8; ++e)
        Cs[wr * 32 + mi * 16 + hh * 8 + e][wc * 32 + ni * 16 + nn] = acc[mi][ni][e];
  __syncthreads();

  const int r = tid >> 2;   // row within tile, 0..63
  const int j = tid & 3;    // 4 threads per row
  const int m = m0 + r;
  const int b = m >> 11;          // / L_
  const int l = m & (L_ - 1);

  if (n0 < NH_ * HD_) {           // ---- Q head: rmsnorm + rope ----
    const int hq = n0 >> 7;
    __bf16* dst = Qo + ((size_t)(b * NH_ + hq) * L_ + l) * HD_;
    float ss = 0.f;
#pragma unroll
    for (int c = 0; c < 32; ++c) { float v = Cs[r][j * 32 + c]; ss += v * v; }
    ss += __shfl_xor(ss, 1, 4);
    ss += __shfl_xor(ss, 2, 4);
    const float rr = rsqrtf(ss * (1.0f / HD_) + 1e-6f);
    const float* cp = cosp + ((size_t)b * L_ + l) * HD_;
    const float* sp = sinp + ((size_t)b * L_ + l) * HD_;
#pragma unroll
    for (int t = 0; t < 16; ++t) {
      int d = j * 16 + t;
      float v1 = Cs[r][d] * rr * qw[d];
      float v2 = Cs[r][d + 64] * rr * qw[d + 64];
      float c = cp[d], s = sp[d];
      dst[d]      = (__bf16)(v1 * c - v2 * s);
      dst[d + 64] = (__bf16)(v2 * c + v1 * s);
    }
  } else if (n0 < NH_ * HD_ + NKV_ * HD_) {   // ---- K head ----
    const int hk = (n0 - NH_ * HD_) >> 7;
    __bf16* dst = Ko + ((size_t)(b * NKV_ + hk) * L_ + l) * HD_;
    float ss = 0.f;
#pragma unroll
    for (int c = 0; c < 32; ++c) { float v = Cs[r][j * 32 + c]; ss += v * v; }
    ss += __shfl_xor(ss, 1, 4);
    ss += __shfl_xor(ss, 2, 4);
    const float rr = rsqrtf(ss * (1.0f / HD_) + 1e-6f);
    const float* cp = cosp + ((size_t)b * L_ + l) * HD_;
    const float* sp = sinp + ((size_t)b * L_ + l) * HD_;
#pragma unroll
    for (int t = 0; t < 16; ++t) {
      int d = j * 16 + t;
      float v1 = Cs[r][d] * rr * kw[d];
      float v2 = Cs[r][d + 64] * rr * kw[d + 64];
      float c = cp[d], s = sp[d];
      dst[d]      = (__bf16)(v1 * c - v2 * s);
      dst[d + 64] = (__bf16)(v2 * c + v1 * s);
    }
  } else {                                    // ---- V head: store transposed [b][hv][d][l] ----
    const int hv = (n0 - NH_ * HD_ - NKV_ * HD_) >> 7;
    __bf16* dst = Vto + (size_t)(b * NKV_ + hv) * HD_ * L_;
#pragma unroll
    for (int c = 0; c < 32; ++c) {
      int d = j * 32 + c;
      dst[(size_t)d * L_ + l] = (__bf16)Cs[r][d];
    }
  }
}

// ---------------- flash attention (hybrid full/causal mask) ----------------
// one block = (b, h, 64-query tile); 8 waves (4x2). Double-buffered K/V^T
// tiles staged with async-to-LDS copies.
__global__ __launch_bounds__(256) void attn_kernel(
    const __bf16* __restrict__ Qb, const __bf16* __restrict__ Kb,
    const __bf16* __restrict__ Vtb, __bf16* __restrict__ Ob) {
  __shared__ __align__(16) __bf16 Qs[64][136];
  __shared__ __align__(16) __bf16 Ks[2][64][136];
  __shared__ __align__(16) __bf16 Vts[2][128][72];
  __shared__ __align__(16) __bf16 Ps[64][72];
  __shared__ float row_m[64], row_alpha[64], row_l[64];
  __shared__ float pmax[64][2], psum[64][2];

  const int tid = threadIdx.x, lane = tid & 31, w = tid >> 5;
  const int wr = w >> 1;   // 0..3 : 16 query rows
  const int wc = w & 1;    // 0..1 : 32 key cols (S) / 64 dims (PV)
  const int hh = lane >> 4, nn = lane & 15;

  const int qt = blockIdx.x & 31;
  const int bh = blockIdx.x >> 5;
  const int b = bh / NH_, h = bh % NH_;
  const int hk = h >> 1;           // NH/NKV = 2
  const int q0 = qt * 64;

  const __bf16* Qg  = Qb  + ((size_t)(b * NH_ + h) * L_ + q0) * HD_;
  const __bf16* Kg0 = Kb  + (size_t)(b * NKV_ + hk) * L_ * HD_;
  const __bf16* Vg0 = Vtb + (size_t)(b * NKV_ + hk) * HD_ * L_;

  // async stage K tile (4 chunks/thread) + V^T tile (4 chunks/thread)
  auto issue_kv = [&](int buf, int kt) {
#pragma unroll
    for (int i = 0; i < 4; ++i) {
      int idx = tid + 256 * i;
      int r = idx >> 4, c = (idx & 15) * 8;
      async_copy16(&Ks[buf][r][c], Kg0 + (size_t)(kt * 64 + r) * HD_ + c);
    }
#pragma unroll
    for (int i = 0; i < 4; ++i) {
      int idx = tid + 256 * i;
      int r = idx >> 3, c = (idx & 7) * 8;
      async_copy16(&Vts[buf][r][c], Vg0 + (size_t)r * L_ + kt * 64 + c);
    }
  };

#pragma unroll
  for (int i = 0; i < 4; ++i) {               // Q tile 64x128 (async)
    int idx = tid + 256 * i;
    int r = idx >> 4, c = (idx & 15) * 8;
    async_copy16(&Qs[r][c], Qg + (size_t)r * HD_ + c);
  }
  issue_kv(0, 0);
  if (tid < 64) { row_m[tid] = -1e30f; row_l[tid] = 0.f; }

  v8f O[4] = {};
  const bool causal = (q0 >= NIMG_);
  const int ktEnd = causal ? (qt + 1) : (L_ / 64);

  for (int kt = 0; kt < ktEnd; ++kt) {
    if (kt + 1 < ktEnd) {
      issue_kv((kt + 1) & 1, kt + 1);
      S_WAIT_ASYNC("0x8");          // newest 8 (next tile) may remain in flight
    } else {
      S_WAIT_ASYNC("0x0");
    }
    __syncthreads();                 // B1: Q + tile kt resident
    const int bi = kt & 1;

    // S = Q K^T / sqrt(d)
    v8f s[2] = {};
#pragma unroll
    for (int d0 = 0; d0 < 128; d0 += 32) {
      v16bf a = lds_load_a(&Qs[0][0], wr * 16, 136, d0);
#pragma unroll
      for (int ni = 0; ni < 2; ++ni) {
        v16bf bb = lds_load_b(&Ks[bi][0][0], wc * 32 + ni * 16, 136, d0);
        s[ni] = __builtin_amdgcn_wmma_f32_16x16x32_bf16(false, a, false, bb, (short)0,
                                                        s[ni], false, false);
      }
    }
    const float scale = 0.08838834764831845f;  // 1/sqrt(128)
#pragma unroll
    for (int ni = 0; ni < 2; ++ni)
#pragma unroll
      for (int e = 0; e < 8; ++e) s[ni][e] *= scale;

    if (causal && kt == qt) {                  // diagonal tile: elementwise causal mask
#pragma unroll
      for (int ni = 0; ni < 2; ++ni)
#pragma unroll
        for (int e = 0; e < 8; ++e) {
          int row = wr * 16 + hh * 8 + e;
          int col = wc * 32 + ni * 16 + nn;
          if (col > row) s[ni][e] = -3.0e38f;
        }
    }

    // partial row max over this wave's 32 key cols
    float rm[8];
#pragma unroll
    for (int e = 0; e < 8; ++e) rm[e] = fmaxf(s[0][e], s[1][e]);
#pragma unroll
    for (int off = 1; off < 16; off <<= 1)
#pragma unroll
      for (int e = 0; e < 8; ++e) rm[e] = fmaxf(rm[e], __shfl_xor(rm[e], off, 16));
    if (nn == 0) {
#pragma unroll
      for (int e = 0; e < 8; ++e) pmax[wr * 16 + hh * 8 + e][wc] = rm[e];
    }
    __syncthreads();                 // B2
    if (tid < 64) {                  // combine stats, update m, compute alpha
      float mo = row_m[tid];
      float mn = fmaxf(mo, fmaxf(pmax[tid][0], pmax[tid][1]));
      row_alpha[tid] = __expf(mo - mn);
      row_m[tid] = mn;
    }
    __syncthreads();                 // B3

    // P = exp(S - m_new), partial row sums, stage P in LDS
    float rs[8] = {};
#pragma unroll
    for (int e = 0; e < 8; ++e) {
      int row = wr * 16 + hh * 8 + e;
      float mn = row_m[row];
#pragma unroll
      for (int ni = 0; ni < 2; ++ni) {
        float p = __expf(s[ni][e] - mn);
        rs[e] += p;
        Ps[row][wc * 32 + ni * 16 + nn] = (__bf16)p;
      }
    }
#pragma unroll
    for (int off = 1; off < 16; off <<= 1)
#pragma unroll
      for (int e = 0; e < 8; ++e) rs[e] += __shfl_xor(rs[e], off, 16);
    if (nn == 0) {
#pragma unroll
      for (int e = 0; e < 8; ++e) psum[wr * 16 + hh * 8 + e][wc] = rs[e];
    }
    __syncthreads();                 // B4

    // O = O*alpha + P @ V
    float al[8];
#pragma unroll
    for (int e = 0; e < 8; ++e) al[e] = row_alpha[wr * 16 + hh * 8 + e];
#pragma unroll
    for (int ni = 0; ni < 4; ++ni)
#pragma unroll
      for (int e = 0; e < 8; ++e) O[ni][e] *= al[e];
#pragma unroll
    for (int k0 = 0; k0 < 64; k0 += 32) {
      v16bf a = lds_load_a(&Ps[0][0], wr * 16, 72, k0);
#pragma unroll
      for (int ni = 0; ni < 4; ++ni) {
        v16bf bb = lds_load_b(&Vts[bi][0][0], wc * 64 + ni * 16, 72, k0);
        O[ni] = __builtin_amdgcn_wmma_f32_16x16x32_bf16(false, a, false, bb, (short)0,
                                                        O[ni], false, false);
      }
    }
    if (tid < 64) row_l[tid] = row_l[tid] * row_alpha[tid] + psum[tid][0] + psum[tid][1];
    __syncthreads();                 // B5: buf bi / Ps free for reuse
  }

  float inv[8];
#pragma unroll
  for (int e = 0; e < 8; ++e) inv[e] = 1.0f / row_l[wr * 16 + hh * 8 + e];
#pragma unroll
  for (int ni = 0; ni < 4; ++ni)
#pragma unroll
    for (int e = 0; e < 8; ++e) {
      int qrow = q0 + wr * 16 + hh * 8 + e;
      int col = h * HD_ + wc * 64 + ni * 16 + nn;
      Ob[((size_t)b * L_ + qrow) * (NH_ * HD_) + col] = (__bf16)(O[ni][e] * inv[e]);
    }
}

// ---------------- output projection: out = attn @ Wo^T (fp32 out) ----------------
__global__ __launch_bounds__(256) void out_gemm_kernel(
    const __bf16* __restrict__ ab, const __bf16* __restrict__ wb,
    float* __restrict__ out) {
  __shared__ __align__(16) __bf16 As[2][64][72];
  __shared__ __align__(16) __bf16 Bs[2][128][72];

  const int tid = threadIdx.x;
  const int lane = tid & 31;
  const int w  = tid >> 5;
  const int wr = w >> 2, wc = w & 3;
  const int hh = lane >> 4, nn = lane & 15;

  const int m0 = blockIdx.y * 64;
  const int n0 = blockIdx.x * 128;

  v8f acc[2][2] = {};

  auto issue_tile = [&](int buf, int k0) {
#pragma unroll
    for (int j = 0; j < 2; ++j) {
      int idx = tid + 256 * j;
      int r = idx >> 3, c = (idx & 7) * 8;
      async_copy16(&As[buf][r][c], ab + (size_t)(m0 + r) * HID_ + k0 + c);
    }
#pragma unroll
    for (int j = 0; j < 4; ++j) {
      int idx = tid + 256 * j;
      int r = idx >> 3, c = (idx & 7) * 8;
      async_copy16(&Bs[buf][r][c], wb + (size_t)(n0 + r) * HID_ + k0 + c);
    }
  };

  issue_tile(0, 0);
  const int NT = HID_ / 64;
  for (int t = 0; t < NT; ++t) {
    if (t + 1 < NT) {
      issue_tile((t + 1) & 1, (t + 1) * 64);
      S_WAIT_ASYNC("0x6");
    } else {
      S_WAIT_ASYNC("0x0");
    }
    __syncthreads();
    const int bi = t & 1;
#pragma unroll
    for (int kk = 0; kk < 64; kk += 32) {
      v16bf afr[2], bfr[2];
#pragma unroll
      for (int mi = 0; mi < 2; ++mi)
        afr[mi] = lds_load_a(&As[bi][0][0], wr * 32 + mi * 16, 72, kk);
#pragma unroll
      for (int ni = 0; ni < 2; ++ni)
        bfr[ni] = lds_load_b(&Bs[bi][0][0], wc * 32 + ni * 16, 72, kk);
#pragma unroll
      for (int mi = 0; mi < 2; ++mi)
#pragma unroll
        for (int ni = 0; ni < 2; ++ni)
          acc[mi][ni] = __builtin_amdgcn_wmma_f32_16x16x32_bf16(
              false, afr[mi], false, bfr[ni], (short)0, acc[mi][ni], false, false);
    }
    __syncthreads();
  }

#pragma unroll
  for (int mi = 0; mi < 2; ++mi)
#pragma unroll
    for (int ni = 0; ni < 2; ++ni)
#pragma unroll
      for (int e = 0; e < 8; ++e) {
        int row = m0 + wr * 32 + mi * 16 + hh * 8 + e;
        int col = n0 + wc * 32 + ni * 16 + nn;
        out[(size_t)row * HID_ + col] = acc[mi][ni][e];
      }
}

// ---------------- launcher ----------------
extern "C" void kernel_launch(void* const* d_in, const int* in_sizes, int n_in,
                              void* d_out, int out_size, void* d_ws, size_t ws_size,
                              hipStream_t stream) {
  (void)in_sizes; (void)n_in; (void)out_size; (void)ws_size;
  const float* x    = (const float*)d_in[0];
  const float* cosp = (const float*)d_in[1];
  const float* sinp = (const float*)d_in[2];
  // d_in[3] = attention_mask (handled analytically)
  const float* Wq = (const float*)d_in[4];
  const float* Wk = (const float*)d_in[5];
  const float* Wv = (const float*)d_in[6];
  const float* Wo = (const float*)d_in[7];
  const float* qw = (const float*)d_in[8];
  const float* kw = (const float*)d_in[9];
  float* out = (float*)d_out;

  char* ws = (char*)d_ws;
  __bf16* xb    = (__bf16*)(ws);                          // 16 MB  x  (bf16)
  __bf16* wqkvb = (__bf16*)(ws + (size_t)16 * (1 << 20)); // 16 MB  Wq|Wk|Wv (bf16)
  __bf16* wob   = (__bf16*)(ws + (size_t)32 * (1 << 20)); //  8 MB  Wo (bf16)
  __bf16* Qb    = (__bf16*)(ws + (size_t)40 * (1 << 20)); // 16 MB  Q  [B,NH,L,HD]
  __bf16* Kbuf  = (__bf16*)(ws + (size_t)56 * (1 << 20)); //  8 MB  K  [B,NKV,L,HD]
  __bf16* Vtb   = (__bf16*)(ws + (size_t)64 * (1 << 20)); //  8 MB  V^T[B,NKV,HD,L]
  __bf16* attnb = (__bf16*)(ws + (size_t)72 * (1 << 20)); // 16 MB  attn out [B*L,2048]

  auto cvt = [&](const float* s, __bf16* d, size_t nelem) {
    int nvec = (int)(nelem / 4);
    cvt_f32_bf16<<<(nvec + 255) / 256, 256, 0, stream>>>((const float4*)s, (bf4*)d, nvec);
  };
  cvt(x,  xb,    (size_t)B_ * L_ * HID_);
  cvt(Wq, wqkvb,                        (size_t)NH_ * HD_ * HID_);
  cvt(Wk, wqkvb + (size_t)2048 * HID_,  (size_t)NKV_ * HD_ * HID_);
  cvt(Wv, wqkvb + (size_t)3072 * HID_,  (size_t)NKV_ * HD_ * HID_);
  cvt(Wo, wob,   (size_t)HID_ * HID_);

  qkv_gemm_kernel<<<dim3(32, 64), 256, 0, stream>>>(xb, wqkvb, cosp, sinp, qw, kw,
                                                    Qb, Kbuf, Vtb);
  attn_kernel<<<dim3(B_ * NH_ * (L_ / 64)), 256, 0, stream>>>(Qb, Kbuf, Vtb, attnb);
  out_gemm_kernel<<<dim3(16, 64), 256, 0, stream>>>(attnb, wob, out);
}
